// UniTransformer_89713276879185
// MI455X (gfx1250) — matrix operand order repaired and checked by reading, not compiled, not run on hardware
//
#include <hip/hip_runtime.h>
#include <hip/hip_bf16.h>

// ---------------- problem constants ----------------
#define NN   4096      // nodes
#define EE   65536     // edges
#define HID  128
#define NH   16
#define HD   8
#define KVD  340
#define KVP  352       // padded K (11 * 32)
#define NL   4

typedef float    v8f  __attribute__((ext_vector_type(8)));
typedef _Float16 v16h __attribute__((ext_vector_type(16)));
typedef _Float16 v8h  __attribute__((ext_vector_type(8)));

__constant__ float c_offset[20] = {0.f,1.f,1.25f,1.5f,1.75f,2.f,2.25f,2.5f,2.75f,3.f,
                                   3.5f,4.f,4.5f,5.f,5.5f,6.f,7.f,8.f,9.f,10.f};

__device__ __forceinline__ v8f zero8() {
  v8f z = {0.f,0.f,0.f,0.f,0.f,0.f,0.f,0.f};
  return z;
}

// Build a 16-half fragment from two aligned 8-half chunks.
__device__ __forceinline__ v16h ld_frag(const _Float16* p0, const _Float16* p1) {
  v8h a = *(const v8h*)p0;
  v8h b = *(const v8h*)p1;
  return __builtin_shufflevector(a, b, 0,1,2,3,4,5,6,7,8,9,10,11,12,13,14,15);
}

__device__ __forceinline__ v8f wmma_f16(v16h a, v16h b, v8f c) {
  return __builtin_amdgcn_wmma_f32_16x16x32_f16(false, a, false, b, (short)0, c, false, false);
}

// monotone float ordering in unsigned space (for atomicMax on floats)
__device__ __forceinline__ unsigned enc_f(float f) {
  unsigned u = __float_as_uint(f);
  return (u & 0x80000000u) ? ~u : (u | 0x80000000u);
}
__device__ __forceinline__ float dec_f(unsigned u) {
  return __uint_as_float((u & 0x80000000u) ? (u & 0x7FFFFFFFu) : ~u);
}

// ============ GEMM1: out16[rows,128] = relu(LN(A[rows,K]@W1 + b1)*g + be) ============
// A fp16 [rows, lda]; Wt fp16 [128][ldw] (W1 transposed, K zero-padded to ldw).
// Block: 128 threads = 4 waves; each block does one 16-row tile; wave w covers cols [32w,32w+32).
__global__ void __launch_bounds__(128)
k_gemm_ln_relu(const _Float16* __restrict__ A, int lda, int ksteps,
               const _Float16* __restrict__ Wt, int ldw,
               const float* __restrict__ b1, const float* __restrict__ gam,
               const float* __restrict__ bet, _Float16* __restrict__ out)
{
  __shared__ float stile[16][132];
  __shared__ float s_mean[16], s_rstd[16];
  const int row0 = blockIdx.x * 16;
  const int lane = threadIdx.x & 31;
  const int wave = threadIdx.x >> 5;
  const int gl = lane >> 4;     // 0/1 half-wave
  const int l16 = lane & 15;
  const int n0 = wave * 32;

  v8f acc0 = zero8(), acc1 = zero8();
  const _Float16* arow  = A  + (size_t)(row0 + l16) * lda;
  const _Float16* brow0 = Wt + (size_t)(n0 + l16) * ldw;
  const _Float16* brow1 = Wt + (size_t)(n0 + 16 + l16) * ldw;
  for (int ks = 0; ks < ksteps; ++ks) {
    const int k0 = ks * 32;
    v16h af  = ld_frag(arow + k0 + gl*16, arow + k0 + gl*16 + 8);
    v16h bf0 = ld_frag(brow0 + k0 + gl*16, brow0 + k0 + gl*16 + 8);
    v16h bf1 = ld_frag(brow1 + k0 + gl*16, brow1 + k0 + gl*16 + 8);
    acc0 = wmma_f16(af, bf0, acc0);
    acc1 = wmma_f16(af, bf1, acc1);
  }
#pragma unroll
  for (int i = 0; i < 8; ++i) {
    const int m = gl*8 + i;
    stile[m][n0 + l16]      = acc0[i] + b1[n0 + l16];
    stile[m][n0 + 16 + l16] = acc1[i] + b1[n0 + 16 + l16];
  }
  __syncthreads();
  if (threadIdx.x < 16) {
    float s = 0.f, s2 = 0.f;
    for (int c = 0; c < 128; ++c) { float v = stile[threadIdx.x][c]; s += v; s2 += v*v; }
    float m = s * (1.f/128.f);
    float var = s2 * (1.f/128.f) - m*m;
    s_mean[threadIdx.x] = m;
    s_rstd[threadIdx.x] = rsqrtf(var + 1e-5f);
  }
  __syncthreads();
  const int c = threadIdx.x;
  const float gc = gam[c], bc = bet[c];
#pragma unroll 4
  for (int r = 0; r < 16; ++r) {
    float v = (stile[r][c] - s_mean[r]) * s_rstd[r] * gc + bc;
    out[(size_t)(row0 + r) * 128 + c] = (_Float16)(v > 0.f ? v : 0.f);
  }
}

// ============ GEMM2 (Nout=128): acc = A[rows,128]@W2 + b2, epilogue by mode ============
// mode 0: outH fp16 = acc        (attention keys)
// mode 1: outF fp32 = acc*ew[row](values scaled by edge weight)
// mode 2: outF fp32 = acc        (queries)
// mode 3: outF fp32 = acc+resid  (output MLP residual)
__global__ void __launch_bounds__(128)
k_gemm2_n128(const _Float16* __restrict__ A, const _Float16* __restrict__ Wt,
             const float* __restrict__ b2, int mode,
             const float* __restrict__ ew, const float* __restrict__ resid,
             float* __restrict__ outF, _Float16* __restrict__ outH)
{
  const int row0 = blockIdx.x * 16;
  const int lane = threadIdx.x & 31;
  const int wave = threadIdx.x >> 5;
  const int gl = lane >> 4, l16 = lane & 15;
  const int n0 = wave * 32;
  const _Float16* arow  = A  + (size_t)(row0 + l16) * 128;
  const _Float16* brow0 = Wt + (size_t)(n0 + l16) * 128;
  const _Float16* brow1 = Wt + (size_t)(n0 + 16 + l16) * 128;
  v8f acc0 = zero8(), acc1 = zero8();
#pragma unroll
  for (int ks = 0; ks < 4; ++ks) {
    const int k0 = ks * 32;
    v16h af  = ld_frag(arow + k0 + gl*16, arow + k0 + gl*16 + 8);
    v16h bf0 = ld_frag(brow0 + k0 + gl*16, brow0 + k0 + gl*16 + 8);
    v16h bf1 = ld_frag(brow1 + k0 + gl*16, brow1 + k0 + gl*16 + 8);
    acc0 = wmma_f16(af, bf0, acc0);
    acc1 = wmma_f16(af, bf1, acc1);
  }
#pragma unroll
  for (int i = 0; i < 8; ++i) {
    const int m  = row0 + gl*8 + i;
    const int c0 = n0 + l16, c1 = n0 + 16 + l16;
    float v0 = acc0[i] + b2[c0];
    float v1 = acc1[i] + b2[c1];
    if (mode == 0) {
      outH[(size_t)m*128 + c0] = (_Float16)v0;
      outH[(size_t)m*128 + c1] = (_Float16)v1;
    } else if (mode == 1) {
      float w = ew[m];
      outF[(size_t)m*128 + c0] = v0 * w;
      outF[(size_t)m*128 + c1] = v1 * w;
    } else if (mode == 2) {
      outF[(size_t)m*128 + c0] = v0;
      outF[(size_t)m*128 + c1] = v1;
    } else {
      outF[(size_t)m*128 + c0] = v0 + resid[(size_t)m*128 + c0];
      outF[(size_t)m*128 + c1] = v1 + resid[(size_t)m*128 + c1];
    }
  }
}

// ============ GEMM2 (Nout=16): outF[rows,16] = (A@W2 + b2) * ew[row] ============
// Block 128 threads = 4 waves; each wave owns a 16-row tile -> 64 rows/block.
__global__ void __launch_bounds__(128)
k_gemm2_n16(const _Float16* __restrict__ A, const _Float16* __restrict__ Wt,
            const float* __restrict__ b2, const float* __restrict__ ew,
            float* __restrict__ outF)
{
  const int lane = threadIdx.x & 31;
  const int wave = threadIdx.x >> 5;
  const int gl = lane >> 4, l16 = lane & 15;
  const int row0 = blockIdx.x * 64 + wave * 16;
  const _Float16* arow = A  + (size_t)(row0 + l16) * 128;
  const _Float16* brow = Wt + (size_t)l16 * 128;
  v8f acc = zero8();
#pragma unroll
  for (int ks = 0; ks < 4; ++ks) {
    const int k0 = ks * 32;
    v16h af = ld_frag(arow + k0 + gl*16, arow + k0 + gl*16 + 8);
    v16h bf = ld_frag(brow + k0 + gl*16, brow + k0 + gl*16 + 8);
    acc = wmma_f16(af, bf, acc);
  }
#pragma unroll
  for (int i = 0; i < 8; ++i) {
    const int m = row0 + gl*8 + i;
    outF[(size_t)m*16 + l16] = (acc[i] + b2[l16]) * ew[m];
  }
}

// ============ small elementwise / graph kernels ============

// per edge: edge_attr one-hot, gaussian dist features into kv cols [0,84), zero pad cols [340,352)
__global__ void k_edge_geom(const float* __restrict__ x, const int* __restrict__ ml,
                            const int* __restrict__ src, const int* __restrict__ dst,
                            _Float16* __restrict__ kvh, float* __restrict__ relx)
{
  int e = blockIdx.x * blockDim.x + threadIdx.x;
  if (e >= EE) return;
  int s = src[e], d = dst[e];
  float rx = x[d*3+0] - x[s*3+0];
  float ry = x[d*3+1] - x[s*3+1];
  float rz = x[d*3+2] - x[s*3+2];
  relx[e*3+0] = rx; relx[e*3+1] = ry; relx[e*3+2] = rz;
  float dist = sqrtf(rx*rx + ry*ry + rz*rz);
  int et = ml[s]*2 + ml[d];
  _Float16* kp = kvh + (size_t)e * KVP;
#pragma unroll
  for (int t = 0; t < 4; ++t) kp[t] = (t == et) ? (_Float16)1.f : (_Float16)0.f;
  for (int j = 0; j < 20; ++j) {
    float dd = dist - c_offset[j];
    float gg = expf(-0.5f * dd * dd);
#pragma unroll
    for (int t = 0; t < 4; ++t)
      kp[4 + t*20 + j] = (t == et) ? (_Float16)gg : (_Float16)0.f;
  }
#pragma unroll
  for (int j = KVD; j < KVP; ++j) kp[j] = (_Float16)0.f;
}

// kv cols [84,340) = fp16(h[dst]) ++ fp16(h[src])
__global__ void k_gather_h(const float* __restrict__ h, const int* __restrict__ src,
                           const int* __restrict__ dst, _Float16* __restrict__ kvh)
{
  int idx = blockIdx.x * blockDim.x + threadIdx.x;   // e*256 + c
  int e = idx >> 8, c = idx & 255;
  int node = (c < 128) ? dst[e] : src[e];
  kvh[(size_t)e * KVP + 84 + c] = (_Float16)h[(size_t)node * 128 + (c & 127)];
}

// ew[e] = sigmoid(df . W + b), df = kv cols [4,84)
__global__ void k_compute_ew(const _Float16* __restrict__ kvh, const float* __restrict__ W,
                             const float* __restrict__ b, float* __restrict__ ew)
{
  int e = blockIdx.x * blockDim.x + threadIdx.x;
  if (e >= EE) return;
  const _Float16* dfp = kvh + (size_t)e * KVP + 4;
  float s = b[0];
  for (int j = 0; j < 80; ++j) s += (float)dfp[j] * W[j];
  ew[e] = 1.f / (1.f + expf(-s));
}

__global__ void k_f32tof16(const float* __restrict__ in, _Float16* __restrict__ out, int n)
{
  int i = blockIdx.x * blockDim.x + threadIdx.x;
  if (i < n) out[i] = (_Float16)in[i];
}

// hcat[i, 0:128] = agg[i], hcat[i, 128:256] = h[i]
__global__ void k_build_hcat(const float* __restrict__ agg, const float* __restrict__ h,
                             _Float16* __restrict__ hcat)
{
  int idx = blockIdx.x * blockDim.x + threadIdx.x;  // i*256 + c
  int i = idx >> 8, c = idx & 255;
  float v = (c < 128) ? agg[(size_t)i*128 + c] : h[(size_t)i*128 + (c - 128)];
  hcat[(size_t)i*256 + c] = (_Float16)v;
}

// logits[e,h] = dot(q[dst[e],h,:], k[e,h,:]) / sqrt(8)
__global__ void k_logits(const float* __restrict__ q, const _Float16* __restrict__ kE,
                         const int* __restrict__ dst, float* __restrict__ logits)
{
  int idx = blockIdx.x * blockDim.x + threadIdx.x;  // e*16 + h
  int e = idx >> 4, hh = idx & 15;
  int d = dst[e];
  const float*    qp = q  + (size_t)d*128 + hh*8;
  const _Float16* kp = kE + (size_t)e*128 + hh*8;
  float s = 0.f;
#pragma unroll
  for (int j = 0; j < 8; ++j) s += qp[j] * (float)kp[j];
  logits[idx] = s * 0.35355339059327373f;
}

__global__ void k_segmax(const float* __restrict__ logits, const int* __restrict__ dst,
                         unsigned* __restrict__ segmax)
{
  int idx = blockIdx.x * blockDim.x + threadIdx.x;
  int e = idx >> 4, hh = idx & 15;
  atomicMax(&segmax[dst[e]*16 + hh], enc_f(logits[idx]));
}

__global__ void k_expden(float* __restrict__ logits, const int* __restrict__ dst,
                         const unsigned* __restrict__ segmax, float* __restrict__ segden)
{
  int idx = blockIdx.x * blockDim.x + threadIdx.x;
  int e = idx >> 4, hh = idx & 15;
  float a = expf(logits[idx] - dec_f(segmax[dst[e]*16 + hh]));
  logits[idx] = a;
  atomicAdd(&segden[dst[e]*16 + hh], a);
}

// agg[dst, h*8+j] += alpha * v[e, h*8+j]
__global__ void k_agg_h(const float* __restrict__ logits, const int* __restrict__ dst,
                        const float* __restrict__ segden, const float* __restrict__ vE,
                        float* __restrict__ agg)
{
  int idx = blockIdx.x * blockDim.x + threadIdx.x;
  int e = idx >> 4, hh = idx & 15;
  int d = dst[e];
  float w = logits[idx] / segden[d*16 + hh];
  const float* vp = vE + (size_t)e*128 + hh*8;
  float* ap = agg + (size_t)d*128 + hh*8;
#pragma unroll
  for (int j = 0; j < 8; ++j) atomicAdd(ap + j, w * vp[j]);
}

// dx[dst,c] += alpha * v16[e,h] * relx[e,c] / NH   (mean over heads)
__global__ void k_agg_x(const float* __restrict__ logits, const int* __restrict__ dst,
                        const float* __restrict__ segden, const float* __restrict__ v16,
                        const float* __restrict__ relx, float* __restrict__ dx)
{
  int idx = blockIdx.x * blockDim.x + threadIdx.x;
  int e = idx >> 4, hh = idx & 15;
  int d = dst[e];
  float w = (logits[idx] / segden[d*16 + hh]) * v16[(size_t)e*16 + hh] * (1.f/16.f);
#pragma unroll
  for (int c = 0; c < 3; ++c) atomicAdd(&dx[d*3 + c], w * relx[e*3 + c]);
}

__global__ void k_x_update(const float* __restrict__ dx, const int* __restrict__ ml,
                           float* __restrict__ xbuf)
{
  int idx = blockIdx.x * blockDim.x + threadIdx.x;  // i*3 + c
  if (idx >= NN*3) return;
  int i = idx / 3;
  xbuf[idx] += dx[idx] * (ml[i] ? 1.f : 0.f);
}

// W [K][Nc] fp32 -> Wt [Nc][Kpad] fp16 (K zero-padded)
__global__ void k_wconv(const float* __restrict__ W, _Float16* __restrict__ Wt,
                        int K, int Nc, int Kpad, int total)
{
  int idx = blockIdx.x * blockDim.x + threadIdx.x;
  if (idx >= total) return;
  int n = idx / Kpad, k = idx - n * Kpad;
  Wt[idx] = (k < K) ? (_Float16)W[(size_t)k * Nc + n] : (_Float16)0.f;
}

// ---------------- host orchestration ----------------

struct MlpP { const float *W1, *W2, *b1, *b2, *be, *g; };
struct WB   { _Float16 *W1t, *W2t; };

extern "C" void kernel_launch(void* const* d_in, const int* in_sizes, int n_in,
                              void* d_out, int out_size, void* d_ws, size_t ws_size,
                              hipStream_t stream)
{
  (void)in_sizes; (void)n_in; (void)out_size; (void)ws_size;

  const float* h_in = (const float*)d_in[0];
  const float* x_in = (const float*)d_in[1];
  const int*   mask = (const int*)d_in[2];
  const int*   eidx = (const int*)d_in[3];
  const int*   src  = eidx;
  const int*   dst  = eidx + EE;

  auto P = [&](int i) -> const float* { return (const float*)d_in[i]; };
  auto getMlp = [&](int b) -> MlpP { return MlpP{P(b), P(b+1), P(b+2), P(b+3), P(b+4), P(b+5)}; };

  // bump allocator on workspace
  char* ws = (char*)d_ws;
  size_t off = 0;
  auto alloc = [&](size_t bytes) -> char* {
    char* p = ws + off;
    off += (bytes + 255) & ~(size_t)255;
    return p;
  };
  _Float16* kvh    = (_Float16*)alloc((size_t)EE * KVP * 2);
  _Float16* hidE   = (_Float16*)alloc((size_t)EE * 128 * 2);
  _Float16* kE     = (_Float16*)alloc((size_t)EE * 128 * 2);
  float*    vE     = (float*)   alloc((size_t)EE * 128 * 4);
  float*    logits = (float*)   alloc((size_t)EE * 16 * 4);
  unsigned* segmax = (unsigned*)alloc((size_t)NN * 16 * 4);
  float*    segden = (float*)   alloc((size_t)NN * 16 * 4);
  float*    qbuf   = (float*)   alloc((size_t)NN * 128 * 4);
  float*    agg    = (float*)   alloc((size_t)NN * 128 * 4);
  float*    hbuf   = (float*)   alloc((size_t)NN * 128 * 4);
  float*    xbuf   = (float*)   alloc((size_t)NN * 3 * 4);
  float*    relx   = (float*)   alloc((size_t)EE * 3 * 4);
  float*    ewbuf  = (float*)   alloc((size_t)EE * 4);
  float*    dxbuf  = (float*)   alloc((size_t)NN * 3 * 4);
  _Float16* hh     = (_Float16*)alloc((size_t)NN * 128 * 2);
  _Float16* hcat   = (_Float16*)alloc((size_t)NN * 256 * 2);

  // fp16 transposed weights per layer
  WB whk[NL], whv[NL], whq[NL], wout[NL], wxk[NL], wxv[NL], wxq[NL];
  for (int L = 0; L < NL; ++L) {
    whk[L]  = { (_Float16*)alloc(128*KVP*2), (_Float16*)alloc(128*128*2) };
    whv[L]  = { (_Float16*)alloc(128*KVP*2), (_Float16*)alloc(128*128*2) };
    whq[L]  = { (_Float16*)alloc(128*128*2), (_Float16*)alloc(128*128*2) };
    wout[L] = { (_Float16*)alloc(128*256*2), (_Float16*)alloc(128*128*2) };
    wxk[L]  = { (_Float16*)alloc(128*KVP*2), (_Float16*)alloc(128*128*2) };
    wxv[L]  = { (_Float16*)alloc(128*KVP*2), (_Float16*)alloc(16*128*2)  };
    wxq[L]  = { (_Float16*)alloc(128*128*2), (_Float16*)alloc(128*128*2) };
  }

  auto convW = [&](const float* W, _Float16* Wt, int K, int Nc, int Kpad) {
    int total = Nc * Kpad;
    k_wconv<<<(total + 255) / 256, 256, 0, stream>>>(W, Wt, K, Nc, Kpad, total);
  };

  // pytree leaf order (sorted dict keys): per layer -> h2x{ew_W,ew_b,xk,xq,xv}, x2h{ew_W,ew_b,hk,hq,hv,out}
  // each MLP -> {W1,W2,b1,b2,be,g}
  for (int L = 0; L < NL; ++L) {
    int base = 4 + L * 46;
    MlpP xk  = getMlp(base + 2),  xq = getMlp(base + 8),  xv = getMlp(base + 14);
    MlpP hk  = getMlp(base + 22), hq = getMlp(base + 28), hv = getMlp(base + 34);
    MlpP om  = getMlp(base + 40);
    convW(hk.W1, whk[L].W1t, KVD, 128, KVP);  convW(hk.W2, whk[L].W2t, 128, 128, 128);
    convW(hv.W1, whv[L].W1t, KVD, 128, KVP);  convW(hv.W2, whv[L].W2t, 128, 128, 128);
    convW(hq.W1, whq[L].W1t, 128, 128, 128);  convW(hq.W2, whq[L].W2t, 128, 128, 128);
    convW(om.W1, wout[L].W1t, 256, 128, 256); convW(om.W2, wout[L].W2t, 128, 128, 128);
    convW(xk.W1, wxk[L].W1t, KVD, 128, KVP);  convW(xk.W2, wxk[L].W2t, 128, 128, 128);
    convW(xv.W1, wxv[L].W1t, KVD, 128, KVP);  convW(xv.W2, wxv[L].W2t, 128,  16, 128);
    convW(xq.W1, wxq[L].W1t, 128, 128, 128);  convW(xq.W2, wxq[L].W2t, 128, 128, 128);
  }

  hipMemcpyAsync(hbuf, h_in, (size_t)NN*128*4, hipMemcpyDeviceToDevice, stream);
  hipMemcpyAsync(xbuf, x_in, (size_t)NN*3*4,   hipMemcpyDeviceToDevice, stream);

  const int THR = 256;
  for (int L = 0; L < NL; ++L) {
    int base = 4 + L * 46;
    MlpP xk = getMlp(base + 2),  xq = getMlp(base + 8),  xv = getMlp(base + 14);
    MlpP hk = getMlp(base + 22), hq = getMlp(base + 28), hv = getMlp(base + 34);
    MlpP om = getMlp(base + 40);

    k_edge_geom<<<EE/THR, THR, 0, stream>>>(xbuf, mask, src, dst, kvh, relx);

    // ================= x2h (updates h) =================
    k_gather_h<<<(EE*256)/THR, THR, 0, stream>>>(hbuf, src, dst, kvh);
    k_compute_ew<<<EE/THR, THR, 0, stream>>>(kvh, P(base+20), P(base+21), ewbuf);
    // keys
    k_gemm_ln_relu<<<EE/16, 128, 0, stream>>>(kvh, KVP, 11, whk[L].W1t, KVP, hk.b1, hk.g, hk.be, hidE);
    k_gemm2_n128<<<EE/16, 128, 0, stream>>>(hidE, whk[L].W2t, hk.b2, 0, nullptr, nullptr, nullptr, kE);
    // values (scaled by ew)
    k_gemm_ln_relu<<<EE/16, 128, 0, stream>>>(kvh, KVP, 11, whv[L].W1t, KVP, hv.b1, hv.g, hv.be, hidE);
    k_gemm2_n128<<<EE/16, 128, 0, stream>>>(hidE, whv[L].W2t, hv.b2, 1, ewbuf, nullptr, vE, nullptr);
    // queries
    k_f32tof16<<<(NN*128)/THR, THR, 0, stream>>>(hbuf, hh, NN*128);
    k_gemm_ln_relu<<<NN/16, 128, 0, stream>>>(hh, 128, 4, whq[L].W1t, 128, hq.b1, hq.g, hq.be, hidE);
    k_gemm2_n128<<<NN/16, 128, 0, stream>>>(hidE, whq[L].W2t, hq.b2, 2, nullptr, nullptr, qbuf, nullptr);
    // attention
    k_logits<<<(EE*16)/THR, THR, 0, stream>>>(qbuf, kE, dst, logits);
    hipMemsetAsync(segmax, 0, (size_t)NN*16*4, stream);
    hipMemsetAsync(segden, 0, (size_t)NN*16*4, stream);
    k_segmax<<<(EE*16)/THR, THR, 0, stream>>>(logits, dst, segmax);
    k_expden<<<(EE*16)/THR, THR, 0, stream>>>(logits, dst, segmax, segden);
    hipMemsetAsync(agg, 0, (size_t)NN*128*4, stream);
    k_agg_h<<<(EE*16)/THR, THR, 0, stream>>>(logits, dst, segden, vE, agg);
    // output MLP + residual
    k_build_hcat<<<(NN*256)/THR, THR, 0, stream>>>(agg, hbuf, hcat);
    k_gemm_ln_relu<<<NN/16, 128, 0, stream>>>(hcat, 256, 8, wout[L].W1t, 256, om.b1, om.g, om.be, hidE);
    k_gemm2_n128<<<NN/16, 128, 0, stream>>>(hidE, wout[L].W2t, om.b2, 3, nullptr, hbuf, hbuf, nullptr);

    // ================= h2x (updates x) =================
    k_gather_h<<<(EE*256)/THR, THR, 0, stream>>>(hbuf, src, dst, kvh);
    k_compute_ew<<<EE/THR, THR, 0, stream>>>(kvh, P(base+0), P(base+1), ewbuf);
    k_gemm_ln_relu<<<EE/16, 128, 0, stream>>>(kvh, KVP, 11, wxk[L].W1t, KVP, xk.b1, xk.g, xk.be, hidE);
    k_gemm2_n128<<<EE/16, 128, 0, stream>>>(hidE, wxk[L].W2t, xk.b2, 0, nullptr, nullptr, nullptr, kE);
    k_gemm_ln_relu<<<EE/16, 128, 0, stream>>>(kvh, KVP, 11, wxv[L].W1t, KVP, xv.b1, xv.g, xv.be, hidE);
    k_gemm2_n16<<<EE/64, 128, 0, stream>>>(hidE, wxv[L].W2t, xv.b2, ewbuf, vE);  // vE reused as [E,16]
    k_f32tof16<<<(NN*128)/THR, THR, 0, stream>>>(hbuf, hh, NN*128);
    k_gemm_ln_relu<<<NN/16, 128, 0, stream>>>(hh, 128, 4, wxq[L].W1t, 128, xq.b1, xq.g, xq.be, hidE);
    k_gemm2_n128<<<NN/16, 128, 0, stream>>>(hidE, wxq[L].W2t, xq.b2, 2, nullptr, nullptr, qbuf, nullptr);
    k_logits<<<(EE*16)/THR, THR, 0, stream>>>(qbuf, kE, dst, logits);
    hipMemsetAsync(segmax, 0, (size_t)NN*16*4, stream);
    hipMemsetAsync(segden, 0, (size_t)NN*16*4, stream);
    k_segmax<<<(EE*16)/THR, THR, 0, stream>>>(logits, dst, segmax);
    k_expden<<<(EE*16)/THR, THR, 0, stream>>>(logits, dst, segmax, segden);
    hipMemsetAsync(dxbuf, 0, (size_t)NN*3*4, stream);
    k_agg_x<<<(EE*16)/THR, THR, 0, stream>>>(logits, dst, segden, vE, relx, dxbuf);
    k_x_update<<<(NN*3 + THR - 1)/THR, THR, 0, stream>>>(dxbuf, mask, xbuf);
  }

  // outputs: h then x, fp32, concatenated
  hipMemcpyAsync(d_out, hbuf, (size_t)NN*128*4, hipMemcpyDeviceToDevice, stream);
  hipMemcpyAsync((float*)d_out + (size_t)NN*128, xbuf, (size_t)NN*3*4, hipMemcpyDeviceToDevice, stream);
}
